// Decoder_37117107372393
// MI455X (gfx1250) — matrix operand order repaired and checked by reading, not compiled, not run on hardware
//
#include <hip/hip_runtime.h>
#include <hip/hip_bf16.h>
#include <stdint.h>

// ---------------------------------------------------------------------------
// LSTM greedy decoder for MI455X (gfx1250, wave32, WMMA bf16 path)
// ---------------------------------------------------------------------------

typedef __bf16 v16bf __attribute__((ext_vector_type(16)));
typedef float  v8f   __attribute__((ext_vector_type(8)));

#define BATCH  128
#define EMBED  512
#define HIDDEN 512
#define VOCAB  32000
#define TSTEPS 20
#define ZLD    (EMBED + HIDDEN)   // 1024, combined [x | h] row stride
#define G4H    (4 * HIDDEN)       // 2048
#define FC_LDSLD 520              // padded LDS row stride (1040B: 16B-aligned,
                                  // 260 dwords == 4 mod 64 banks -> conflict-free)

// Async global->LDS path (CDNA5): use when toolchain exposes the builtins.
#ifdef __has_builtin
#if __has_builtin(__builtin_amdgcn_global_load_async_to_lds_b128) && \
    __has_builtin(__builtin_amdgcn_s_wait_asynccnt)
#define USE_ASYNC_LDS 1
#endif
#endif

#ifdef USE_ASYNC_LDS
// Types matching the builtin's expected parameters (per hipcc diagnostic):
// param0: int __vector(4) pointer in global (AS1), param1: same in LDS (AS3).
typedef int v4i_ __attribute__((vector_size(4 * sizeof(int))));
typedef __attribute__((address_space(1))) v4i_ as1_v4i;
typedef __attribute__((address_space(3))) v4i_ as3_v4i;
#endif

__device__ __forceinline__ float sigm(float x) { return 1.0f / (1.0f + __expf(-x)); }

// Order-preserving float -> uint mapping (monotone: a<b  <=>  key(a)<key(b))
__device__ __forceinline__ unsigned int orderedBits(float f) {
    unsigned int b = __float_as_uint(f);
    return (b & 0x80000000u) ? ~b : (b | 0x80000000u);
}

// A-fragment: 16x32 bf16, M = lane%16; lanes 0-15 hold K = kb+0..15,
// lanes 16-31 hold K = kb+16..31 (same k order as the B-fragment).
__device__ __forceinline__ v16bf load_a_frag(const __bf16* __restrict__ base,
                                             int ld, int m0, int kb, int lane) {
    const int m     = m0 + (lane & 15);
    const int kbase = (lane >> 4) << 4;          // 0 or 16
    const __bf16* p = base + (size_t)m * ld + kb + kbase;
    v16bf a;
#pragma unroll
    for (int e = 0; e < 16; ++e) a[e] = p[e];
    return a;
}

// B-fragment: 32x16 bf16, N = lane%16; lanes 0-15 hold K = kb+0..15,
// lanes 16-31 hold K = kb+16..31. W row-major [N][K] so B[k][n] = W[n][k].
__device__ __forceinline__ v16bf load_b_frag(const __bf16* __restrict__ w,
                                             int ldw, int n0, int kb, int lane) {
    const int n    = n0 + (lane & 15);
    const int koff = (lane >> 4) << 4;           // 0 or 16
    const __bf16* p = w + (size_t)n * ldw + kb + koff;
    v16bf b;
#pragma unroll
    for (int e = 0; e < 16; ++e) b[e] = p[e];
    return b;
}

// Same fragment shape, sourced from the LDS-staged tile (row stride FC_LDSLD).
__device__ __forceinline__ v16bf load_b_frag_lds(const __bf16* Bt, int kb, int lane) {
    const int r    = lane & 15;
    const int koff = (lane >> 4) << 4;
    const __bf16* p = Bt + r * FC_LDSLD + kb + koff;
    v16bf b;
#pragma unroll
    for (int e = 0; e < 16; ++e) b[e] = p[e];
    return b;
}

// ---------------------------------------------------------------------------
// One-time setup kernels (run every kernel_launch call; deterministic)
// ---------------------------------------------------------------------------

// Wc bf16 [2048][1024]: cols 0-511 from W_ih, cols 512-1023 from W_hh.
// Also bias[n] = b_ih[n] + b_hh[n].
__global__ void build_wc_kernel(const float* __restrict__ Wih,
                                const float* __restrict__ Whh,
                                const float* __restrict__ bih,
                                const float* __restrict__ bhh,
                                __bf16* __restrict__ Wc,
                                float* __restrict__ bias) {
    const int idx = blockIdx.x * blockDim.x + threadIdx.x;
    if (idx < G4H * ZLD) {
        const int n = idx / ZLD, k = idx % ZLD;
        const float w = (k < EMBED) ? Wih[(size_t)n * EMBED + k]
                                    : Whh[(size_t)n * HIDDEN + (k - EMBED)];
        Wc[idx] = (__bf16)w;
    }
    if (idx < G4H) bias[idx] = bih[idx] + bhh[idx];
}

__global__ void cvt_wfc_kernel(const float* __restrict__ src,
                               __bf16* __restrict__ dst) {
    const size_t n = (size_t)VOCAB * HIDDEN;
    for (size_t i = (size_t)blockIdx.x * blockDim.x + threadIdx.x; i < n;
         i += (size_t)gridDim.x * blockDim.x)
        dst[i] = (__bf16)src[i];
}

// z0.x = bf16(features), z0.h = 0, c = 0, argmax keys = 0.
__global__ void init_state_kernel(const float* __restrict__ features,
                                  __bf16* __restrict__ z0,
                                  float* __restrict__ c,
                                  unsigned long long* __restrict__ keys) {
    const int idx = blockIdx.x * blockDim.x + threadIdx.x;
    if (idx < BATCH * ZLD) {
        const int b = idx >> 10, k = idx & (ZLD - 1);
        z0[idx] = (k < EMBED) ? (__bf16)features[(size_t)b * EMBED + k] : (__bf16)0.0f;
    }
    if (idx < BATCH * HIDDEN) c[idx] = 0.0f;
    if (idx < BATCH) keys[idx] = 0ULL;
}

// ---------------------------------------------------------------------------
// Per-step kernel 1: gates GEMM + fused LSTM cell update.
// Grid: 32 blocks (16-wide hidden tiles) x 256 threads (8 waves; wave w owns
// rows m0 = 16w). Each block computes i/f/g/o tiles for its hidden slice over
// K=1024 ([x|h] from zin), then updates c and writes new h (bf16) into zout's
// h region (ping-pong buffer -> no cross-block RAW hazard on h).
// ---------------------------------------------------------------------------
__global__ void gates_cell_kernel(const __bf16* __restrict__ Wc,   // [2048][1024]
                                  const __bf16* __restrict__ zin,  // [128][1024]
                                  __bf16* __restrict__ zout,       // [128][1024]
                                  float* __restrict__ c,           // [128][512]
                                  const float* __restrict__ bias)  // [2048]
{
    const int lane = threadIdx.x & 31;
    const int wave = threadIdx.x >> 5;
    const int m0 = wave * 16;
    const int h0 = blockIdx.x * 16;

    v8f acc_i = {}, acc_f = {}, acc_g = {}, acc_o = {};
    for (int kb = 0; kb < ZLD; kb += 32) {
        const v16bf a = load_a_frag(zin, ZLD, m0, kb, lane);
        v16bf b;
        b = load_b_frag(Wc, ZLD, h0 + 0 * HIDDEN, kb, lane);
        acc_i = __builtin_amdgcn_wmma_f32_16x16x32_bf16(false, a, false, b,
                                                        (short)0, acc_i, false, false);
        b = load_b_frag(Wc, ZLD, h0 + 1 * HIDDEN, kb, lane);
        acc_f = __builtin_amdgcn_wmma_f32_16x16x32_bf16(false, a, false, b,
                                                        (short)0, acc_f, false, false);
        b = load_b_frag(Wc, ZLD, h0 + 2 * HIDDEN, kb, lane);
        acc_g = __builtin_amdgcn_wmma_f32_16x16x32_bf16(false, a, false, b,
                                                        (short)0, acc_g, false, false);
        b = load_b_frag(Wc, ZLD, h0 + 3 * HIDDEN, kb, lane);
        acc_o = __builtin_amdgcn_wmma_f32_16x16x32_bf16(false, a, false, b,
                                                        (short)0, acc_o, false, false);
    }

    // C/D layout: N = lane%16, M = v + 8*(lane/16) within this wave's tile.
    const int hcol = h0 + (lane & 15);
    const int mb = m0 + ((lane >> 4) << 3);
    const float bi = bias[hcol];
    const float bf = bias[HIDDEN + hcol];
    const float bg = bias[2 * HIDDEN + hcol];
    const float bo = bias[3 * HIDDEN + hcol];
#pragma unroll
    for (int v = 0; v < 8; ++v) {
        const int m = mb + v;
        const float ig = sigm(acc_i[v] + bi);
        const float fg = sigm(acc_f[v] + bf);
        const float gg = tanhf(acc_g[v] + bg);
        const float og = sigm(acc_o[v] + bo);
        const float cn = fg * c[(size_t)m * HIDDEN + hcol] + ig * gg;
        c[(size_t)m * HIDDEN + hcol] = cn;
        zout[(size_t)m * ZLD + EMBED + hcol] = (__bf16)(og * tanhf(cn));
    }
}

// ---------------------------------------------------------------------------
// Per-step kernel 2: logits GEMM + fused bias + row argmax.
// Grid: VOCAB/16 = 2000 blocks x 8 waves. The block's 16x512 W_fc tile is
// staged ONCE into LDS (async global->LDS, ASYNCcnt-tracked) and shared by
// all 8 waves, cutting W_fc fetch traffic 8x. Packed key =
// ordered(float)<<32|~idx so u64 max gives max logit with lowest-index tie
// break (jnp.argmax rule).
// ---------------------------------------------------------------------------
__global__ void fc_argmax_kernel(const __bf16* __restrict__ Wfc,  // [32000][512]
                                 const __bf16* __restrict__ z,    // h at col 512
                                 const float* __restrict__ bfc,   // [32000]
                                 unsigned long long* __restrict__ keys) // [128]
{
    __shared__ __bf16 Bt[16 * FC_LDSLD];   // 16 rows x 512 bf16 (+pad) ~16.6 KB

    const int tid  = threadIdx.x;
    const int lane = tid & 31;
    const int wave = tid >> 5;
    const int m0 = wave * 16;
    const int n0 = blockIdx.x * 16;

    // Cooperative stage: 256 threads, 16 threads per tile row, 64B each.
    {
        const int r  = tid >> 4;           // 0..15
        const int cs = (tid & 15) * 32;    // element offset within the row
        const __bf16* gsrc = Wfc + (size_t)(n0 + r) * HIDDEN + cs;
        __bf16* ldst = &Bt[r * FC_LDSLD + cs];
#ifdef USE_ASYNC_LDS
#pragma unroll
        for (int j = 0; j < 4; ++j)
            __builtin_amdgcn_global_load_async_to_lds_b128(
                (as1_v4i*)(gsrc + j * 8), (as3_v4i*)(ldst + j * 8), 0, 0);
        __builtin_amdgcn_s_wait_asynccnt(0);
#else
#pragma unroll
        for (int j = 0; j < 4; ++j)
            ((ulonglong2*)ldst)[j] = ((const ulonglong2*)gsrc)[j];
#endif
    }
    __syncthreads();

    v8f acc = {};
    for (int kb = 0; kb < HIDDEN; kb += 32) {
        const v16bf a = load_a_frag(z, ZLD, m0, EMBED + kb, lane);
        const v16bf b = load_b_frag_lds(Bt, kb, lane);
        acc = __builtin_amdgcn_wmma_f32_16x16x32_bf16(false, a, false, b,
                                                      (short)0, acc, false, false);
    }

    const int n = n0 + (lane & 15);
    const float bias = bfc[n];
    const int mb = m0 + ((lane >> 4) << 3);
#pragma unroll
    for (int v = 0; v < 8; ++v) {
        const float logit = acc[v] + bias;
        unsigned long long key =
            ((unsigned long long)orderedBits(logit) << 32) |
            (unsigned long long)(unsigned int)(~(unsigned int)n);
        // max-reduce over the 16 lanes holding the same row (xor<16 keeps halves)
#pragma unroll
        for (int s = 1; s < 16; s <<= 1) {
            const unsigned long long other = __shfl_xor(key, s, 32);
            if (other > key) key = other;
        }
        if ((lane & 15) == 0)
            atomicMax(&keys[mb + v], key);
    }
}

// ---------------------------------------------------------------------------
// Per-step kernel 3: decode argmax, emit outputs, gather next embedding into
// zout.x (bf16), reset key for the next step.
// ---------------------------------------------------------------------------
__global__ void finalize_kernel(unsigned long long* __restrict__ keys,
                                const float* __restrict__ emb,  // [32000][512]
                                __bf16* __restrict__ zout,
                                float* __restrict__ out, int t) {
    const int b = blockIdx.x;
    const unsigned long long key = keys[b];
    __syncthreads();  // everyone read key before thread 0 resets it
    const unsigned int hi = (unsigned int)(key >> 32);
    const unsigned int bits = (hi & 0x80000000u) ? (hi ^ 0x80000000u) : ~hi;
    const float val = __uint_as_float(bits);
    const int idx = (int)(~(unsigned int)(key & 0xFFFFFFFFu));
    if (threadIdx.x == 0) {
        out[(size_t)b * TSTEPS + t] = (float)idx;                   // ids [B,T]
        out[(size_t)BATCH * TSTEPS + (size_t)b * TSTEPS + t] = val; // vals [B,T]
        keys[b] = 0ULL;
    }
    const float* er = emb + (size_t)idx * EMBED;
    for (int k = threadIdx.x; k < EMBED; k += blockDim.x)
        zout[(size_t)b * ZLD + k] = (__bf16)er[k];
}

// ---------------------------------------------------------------------------
extern "C" void kernel_launch(void* const* d_in, const int* in_sizes, int n_in,
                              void* d_out, int out_size, void* d_ws, size_t ws_size,
                              hipStream_t stream) {
    const float* features = (const float*)d_in[0];
    /* d_in[1] = lengths: unused (scan length is static) */
    const float* emb   = (const float*)d_in[2];
    const float* Wih   = (const float*)d_in[3];
    const float* Whh   = (const float*)d_in[4];
    const float* bih   = (const float*)d_in[5];
    const float* bhh   = (const float*)d_in[6];
    const float* WfcF  = (const float*)d_in[7];
    const float* bfc   = (const float*)d_in[8];
    float* out = (float*)d_out;

    // Workspace carve-out (256B aligned)
    char* ws = (char*)d_ws;
    size_t off = 0;
    auto carve = [&](size_t bytes) -> void* {
        void* p = ws + off;
        off = (off + bytes + 255) & ~(size_t)255;
        return p;
    };
    __bf16* Wc   = (__bf16*)carve((size_t)G4H * ZLD * sizeof(__bf16));      // 4 MB
    __bf16* Wfc  = (__bf16*)carve((size_t)VOCAB * HIDDEN * sizeof(__bf16)); // 32.8 MB
    __bf16* z0   = (__bf16*)carve((size_t)BATCH * ZLD * sizeof(__bf16));
    __bf16* z1   = (__bf16*)carve((size_t)BATCH * ZLD * sizeof(__bf16));
    float*  c    = (float*)carve((size_t)BATCH * HIDDEN * sizeof(float));
    float*  bias = (float*)carve((size_t)G4H * sizeof(float));
    unsigned long long* keys =
        (unsigned long long*)carve((size_t)BATCH * sizeof(unsigned long long));

    // One-time conversions + state init (deterministic each call)
    build_wc_kernel<<<(G4H * ZLD + 255) / 256, 256, 0, stream>>>(Wih, Whh, bih, bhh,
                                                                 Wc, bias);
    cvt_wfc_kernel<<<8192, 256, 0, stream>>>(WfcF, Wfc);
    init_state_kernel<<<(BATCH * ZLD + 255) / 256, 256, 0, stream>>>(features, z0,
                                                                     c, keys);

    __bf16* zin = z0;
    __bf16* zout = z1;
    for (int t = 0; t < TSTEPS; ++t) {
        gates_cell_kernel<<<HIDDEN / 16, 256, 0, stream>>>(Wc, zin, zout, c, bias);
        fc_argmax_kernel<<<VOCAB / 16, 256, 0, stream>>>(Wfc, zout, bfc, keys);
        finalize_kernel<<<BATCH, 128, 0, stream>>>(keys, emb, zout, out, t);
        __bf16* tmp = zin; zin = zout; zout = tmp;
    }
}
